// SoftmaxSelfAttention_72567767433637
// MI455X (gfx1250) — compile-verified, matrix operand order checked by reading, not run on hardware
//
#include <hip/hip_runtime.h>

// ---------------------------------------------------------------------------
// Softmax self-attention for MI455X (gfx1250, wave32, WMMA).
//   B=4, S=2048, D=1024.  All matmuls on v_wmma_f32_16x16x32_bf16.
// Pipeline: cvt -> W^T cvt -> QKV GEMMs -> scores GEMM -> softmax(->bf16 P)
//           -> P@V GEMM.  GEMM: 128x128 block tile, 8 waves, B tile staged in
//           LDS via async-to-LDS (double buffered), A fragment double-buffered
//           in registers, B fragments software-pipelined depth-2 from LDS.
// ---------------------------------------------------------------------------

typedef __bf16 bf16_t;
typedef bf16_t v16bf __attribute__((ext_vector_type(16)));
typedef float  v8f   __attribute__((ext_vector_type(8)));
typedef unsigned short u16;
typedef int v4i __attribute__((vector_size(16)));  // matches builtin param type

#if defined(__has_builtin)
#if __has_builtin(__builtin_amdgcn_global_load_async_to_lds_b128) && \
    __has_builtin(__builtin_amdgcn_s_wait_asynccnt)
#define USE_ASYNC_LDS 1
#endif
#endif
#ifndef USE_ASYNC_LDS
#define USE_ASYNC_LDS 0
#endif

#if USE_ASYNC_LDS
typedef __attribute__((address_space(1))) v4i* gptr_v4i;  // global (AS1)
typedef __attribute__((address_space(3))) v4i* lptr_v4i;  // LDS (AS3)
#endif

__device__ __forceinline__ u16 f2bf(float f) {
  union { float f; unsigned u; } c; c.f = f;
  unsigned u = c.u;
  return (u16)((u + 0x7FFFu + ((u >> 16) & 1u)) >> 16);  // round-to-nearest-even
}

// A/B WMMA fragment: 16 bf16 per lane = 8 VGPRs = two 16-byte segments.
// 16-bit A-matrix 16x32 layout (ISA 7.12.2): for half = lane>>4,
//   elements 0..7  = K (8*half)   .. (8*half+7)
//   elements 8..15 = K (16+8*half)..(16+8*half+7)
union FragAB {
  v16bf v;
  uint4 u[2];
  u16   s[16];
};

__device__ __forceinline__ void loadA16(FragAB& f, const u16* __restrict__ arow,
                                        int k0, int half) {
  f.u[0] = *reinterpret_cast<const uint4*>(arow + k0 + 8 * half);
  f.u[1] = *reinterpret_cast<const uint4*>(arow + k0 + 16 + 8 * half);
}

// LDS B tile: 128 rows x 32 k (bf16), row stride padded to 40 u16 (80 B) so the
// 16 rows read by a half-wave of ds_load_b128 start on 16 distinct bank groups.
constexpr int BPAD  = 40;
constexpr int BTILE = 128 * BPAD;  // u16 elements per buffer

// B fragment for n-subtile j from the staged LDS tile.
__device__ __forceinline__ void loadB_lds(FragAB& f, const u16* lbuf, int j,
                                          int lcol, int half) {
  const u16* lb = lbuf + (j * 16 + lcol) * BPAD;
  f.u[0] = *reinterpret_cast<const uint4*>(lb + 8 * half);
  f.u[1] = *reinterpret_cast<const uint4*>(lb + 16 + 8 * half);
}

// Cooperatively stage Brow[nb..nb+127][k0..k0+31] into lbuf.
// 512 16-byte chunks, 256 threads -> 2 chunks per thread (one async b128 each).
__device__ __forceinline__ void stage_b_tile(const u16* __restrict__ gbase, int ldb,
                                             int k0, u16* lbuf, int tid) {
#pragma unroll
  for (int h = 0; h < 2; ++h) {
    const int c   = tid + h * 256;
    const int row = c >> 2;
    const int col = (c & 3) * 8;
    const u16* g = gbase + (long long)row * ldb + (k0 + col);
    u16* l = lbuf + row * BPAD + col;
#if USE_ASYNC_LDS
    __builtin_amdgcn_global_load_async_to_lds_b128((gptr_v4i)g, (lptr_v4i)l, 0, 0);
#else
    *reinterpret_cast<uint4*>(l) = *reinterpret_cast<const uint4*>(g);
#endif
  }
}

__device__ __forceinline__ void stage_fence_barrier() {
#if USE_ASYNC_LDS
  __builtin_amdgcn_s_wait_asynccnt(0);
#endif
  __syncthreads();
}

// ---------------------------------------------------------------------------
// WMMA GEMM:  C[m][n] = scale * sum_k A[m][k] * Brow[n][k]  (+ bias[n])
//   OUTMODE : 0 = bf16 row-major, 1 = bf16 batch-transposed Vt[b][n][t],
//             2 = fp32 row-major
// Block = 256 threads (8 waves). Block tile = 128(M) x 128(N).
// Wave w owns rows [blockIdx.y*128 + w*16, +16), 8 accumulators over 128 cols.
// ---------------------------------------------------------------------------
template <int OUTMODE>
__global__ __launch_bounds__(256) void gemm_bf16_wmma(
    const u16* __restrict__ Ain, const u16* __restrict__ Brow,
    const float* __restrict__ bias, void* __restrict__ Out,
    int K, int N, int lda, int ldb, int ldo, float scale, int tokS,
    long long aStride, long long bStride, long long oStride) {
  __shared__ __align__(16) u16 smem[2 * BTILE];  // 20 KB double-buffered B tile

  const int tid  = threadIdx.x;
  const int wave = tid >> 5;
  const int lane = tid & 31;
  const int half = lane >> 4;
  const int lcol = lane & 15;

  const int m0 = blockIdx.y * 128 + wave * 16;
  const int nb = blockIdx.x * 128;

  const long long za = (long long)blockIdx.z * aStride;
  const long long zb = (long long)blockIdx.z * bStride;
  const long long zo = (long long)blockIdx.z * oStride;

  const u16* gB   = Brow + zb + (long long)nb * ldb;
  const u16* arow = Ain + za + (long long)(m0 + lcol) * lda;

  v8f zero;
#pragma unroll
  for (int i = 0; i < 8; ++i) zero[i] = 0.0f;
  v8f acc[8];
#pragma unroll
  for (int j = 0; j < 8; ++j) acc[j] = zero;

  FragAB acur, anext;
  stage_b_tile(gB, ldb, 0, smem, tid);  // prefetch first B tile
  loadA16(acur, arow, 0, half);

  const int nk = K / 32;
  for (int it = 0; it < nk; ++it) {
    stage_fence_barrier();  // publish buffer it&1; all waves done reading other buf
    u16* cur = smem + (it & 1) * BTILE;
    u16* nxt = smem + ((it & 1) ^ 1) * BTILE;
    const int k0 = it * 32;
    if (it + 1 < nk) {
      stage_b_tile(gB, ldb, k0 + 32, nxt, tid);  // overlaps with compute below
      loadA16(anext, arow, k0 + 32, half);
    }
    // B fragments: depth-2 software pipeline over 3 rotating buffers so each
    // WMMA only waits on DS loads issued two WMMAs earlier.
    FragAB bfr[3];
    loadB_lds(bfr[0], cur, 0, lcol, half);
    loadB_lds(bfr[1], cur, 1, lcol, half);
#pragma unroll
    for (int j = 0; j < 8; ++j) {
      if (j + 2 < 8) loadB_lds(bfr[(j + 2) % 3], cur, j + 2, lcol, half);
      acc[j] = __builtin_amdgcn_wmma_f32_16x16x32_bf16(
          false, acur.v, false, bfr[j % 3].v, (short)0, acc[j], false, false);
    }
    acur = anext;
  }

  // Epilogue. C/D layout (ISA 7.12.2): lane -> n = lcol, VGPR r -> m = 8*half + r.
#pragma unroll
  for (int j = 0; j < 8; ++j) {
    const int ncol = nb + j * 16 + lcol;
    float badd = 0.0f;
    if (bias) badd = bias[ncol];
#pragma unroll
    for (int r = 0; r < 8; ++r) {
      const int mr  = m0 + 8 * half + r;
      const float v = acc[j][r] * scale + badd;
      if constexpr (OUTMODE == 0) {
        ((u16*)Out)[zo + (long long)mr * ldo + ncol] = f2bf(v);
      } else if constexpr (OUTMODE == 1) {
        const int bidx = mr / tokS;         // batch index
        const int t    = mr - bidx * tokS;  // token within batch
        ((u16*)Out)[((long long)bidx * N + ncol) * (long long)tokS + t] = f2bf(v);
      } else {
        ((float*)Out)[zo + (long long)mr * ldo + ncol] = v;
      }
    }
  }
}

// ---------------------------------------------------------------------------
// fp32 -> bf16 elementwise convert
// ---------------------------------------------------------------------------
__global__ void cvt_bf16(const float* __restrict__ in, u16* __restrict__ out, int n) {
  int i = blockIdx.x * blockDim.x + threadIdx.x;
  if (i < n) out[i] = f2bf(in[i]);
}

// ---------------------------------------------------------------------------
// Transpose + convert: in[R][C] fp32 -> out[C][R] bf16  (LDS tiled)
// ---------------------------------------------------------------------------
__global__ __launch_bounds__(256) void transpose_cvt(const float* __restrict__ in,
                                                     u16* __restrict__ out,
                                                     int R, int C) {
  __shared__ float tile[32][33];
  const int bx = blockIdx.x * 32;
  const int by = blockIdx.y * 32;
  const int tx = threadIdx.x;  // 0..31
  const int ty = threadIdx.y;  // 0..7
#pragma unroll
  for (int j = 0; j < 32; j += 8)
    tile[ty + j][tx] = in[(long long)(by + ty + j) * C + (bx + tx)];
  __syncthreads();
#pragma unroll
  for (int j = 0; j < 32; j += 8)
    out[(long long)(bx + ty + j) * R + (by + tx)] = f2bf(tile[tx][ty + j]);
}

// ---------------------------------------------------------------------------
// Row softmax, fp32 in -> bf16 out. One 256-thread block per row (n <= 2048).
// Exp values live in registers between passes (no fp32 write-back).
// ---------------------------------------------------------------------------
__global__ __launch_bounds__(256) void softmax_to_bf16(const float* __restrict__ sc,
                                                       u16* __restrict__ P, int n) {
  __shared__ float red[256];
  const float* src = sc + (long long)blockIdx.x * n;
  u16* dst = P + (long long)blockIdx.x * n;
  const int tid = threadIdx.x;

  float m = -3.402823466e38f;
  for (int i = tid; i < n; i += 256) m = fmaxf(m, src[i]);
  red[tid] = m;
  __syncthreads();
  for (int st = 128; st > 0; st >>= 1) {
    if (tid < st) red[tid] = fmaxf(red[tid], red[tid + st]);
    __syncthreads();
  }
  m = red[0];
  __syncthreads();

  float ev[8];  // n/256 <= 8 values per thread
  float sum = 0.0f;
  int cnt = 0;
  for (int i = tid; i < n; i += 256, ++cnt) {
    float e = __expf(src[i] - m);
    ev[cnt] = e;
    sum += e;
  }
  red[tid] = sum;
  __syncthreads();
  for (int st = 128; st > 0; st >>= 1) {
    if (tid < st) red[tid] += red[tid + st];
    __syncthreads();
  }
  const float inv = 1.0f / red[0];

  cnt = 0;
  for (int i = tid; i < n; i += 256, ++cnt) dst[i] = f2bf(ev[cnt] * inv);
}

// ---------------------------------------------------------------------------
// Workspace (~141 MB):
//   xb   bf16 [B*S][D]    16.8 MB
//   Wt*  bf16 [D][D] x3    6.3 MB  (weights pre-transposed)
//   Qb   bf16 [B*S][D]    16.8 MB  \ after scores GEMM these 33.5 MB are dead
//   Kb   bf16 [B*S][D]    16.8 MB  / and are reused as P bf16 [B][S][S]
//   Vt   bf16 [B][D][S]   16.8 MB
//   sc   fp32 [B][S][S]   67.1 MB  (raw scores)
// ---------------------------------------------------------------------------
extern "C" void kernel_launch(void* const* d_in, const int* in_sizes, int n_in,
                              void* d_out, int out_size, void* d_ws, size_t ws_size,
                              hipStream_t stream) {
  constexpr int B = 4, S = 2048, D = 1024;
  constexpr long long ND = (long long)B * S * D;  // 8,388,608

  const float* x  = (const float*)d_in[0];
  const float* Wq = (const float*)d_in[1];
  const float* bq = (const float*)d_in[2];
  const float* Wk = (const float*)d_in[3];
  const float* bk = (const float*)d_in[4];
  const float* Wv = (const float*)d_in[5];
  const float* bv = (const float*)d_in[6];
  float* out = (float*)d_out;

  char* p = (char*)d_ws;
  u16* xb  = (u16*)p; p += ND * 2;
  u16* Wtq = (u16*)p; p += (long long)D * D * 2;
  u16* Wtk = (u16*)p; p += (long long)D * D * 2;
  u16* Wtv = (u16*)p; p += (long long)D * D * 2;
  u16* Qb  = (u16*)p; p += ND * 2;
  u16* Kb  = (u16*)p; p += ND * 2;
  u16* Vt  = (u16*)p; p += ND * 2;
  float* sc = (float*)p;        // B*S*S fp32
  u16* Pb = Qb;                 // reuse Qb+Kb region: exactly B*S*S bf16

  dim3 blk(256);

  // 1) x -> bf16
  cvt_bf16<<<(unsigned)((ND + 255) / 256), blk, 0, stream>>>(x, xb, (int)ND);

  // 2) W -> W^T bf16
  dim3 tblk(32, 8), tgrd(D / 32, D / 32);
  transpose_cvt<<<tgrd, tblk, 0, stream>>>(Wq, Wtq, D, D);
  transpose_cvt<<<tgrd, tblk, 0, stream>>>(Wk, Wtk, D, D);
  transpose_cvt<<<tgrd, tblk, 0, stream>>>(Wv, Wtv, D, D);

  // 3) QKV projections: M = 8192, N = K = 1024
  dim3 gqkv(D / 128, (B * S) / 128, 1);
  gemm_bf16_wmma<0><<<gqkv, blk, 0, stream>>>(
      xb, Wtq, bq, Qb, D, D, D, D, D, 1.0f, S, 0, 0, 0);
  gemm_bf16_wmma<0><<<gqkv, blk, 0, stream>>>(
      xb, Wtk, bk, Kb, D, D, D, D, D, 1.0f, S, 0, 0, 0);
  gemm_bf16_wmma<1><<<gqkv, blk, 0, stream>>>(  // V stored transposed per batch
      xb, Wtv, bv, Vt, D, D, D, D, 0, 1.0f, S, 0, 0, 0);

  // 4) scores[b] = Q[b] @ K[b]^T / 32 : M=N=2048, K=1024
  dim3 gsc(S / 128, S / 128, B);
  gemm_bf16_wmma<2><<<gsc, blk, 0, stream>>>(
      Qb, Kb, nullptr, sc, D, S, D, D, S, 0.03125f, S,
      (long long)S * D, (long long)S * D, (long long)S * S);

  // 5) softmax rows -> bf16 P (overwrites dead Qb/Kb region)
  softmax_to_bf16<<<B * S, blk, 0, stream>>>(sc, Pb, S);

  // 6) out[b] = P[b] @ V[b] : M=2048, N=1024, K=2048
  dim3 gav(D / 128, S / 128, B);
  gemm_bf16_wmma<2><<<gav, blk, 0, stream>>>(
      Pb, Vt, nullptr, out, S, D, S, S, D, 1.0f, S,
      (long long)S * S, (long long)D * S, (long long)S * D);
}
